// GraphAttention_50543175139579
// MI455X (gfx1250) — compile-verified, matrix operand order checked by reading, not run on hardware
//
#include <hip/hip_runtime.h>
#include <hip/hip_bf16.h>

typedef __attribute__((ext_vector_type(16))) _Float16 v16h;
typedef __attribute__((ext_vector_type(8)))  float    v8f;
typedef __attribute__((ext_vector_type(4)))  float    v4f;
typedef __attribute__((ext_vector_type(4)))  int      v4i;

#define N_ROWS 8192
#define IN_DIM 512
#define OUT_DIM 64
#define ALPHA 0.2f
#define NEGBIG (-1.0e30f)
#define JSPLIT 8                      // j-partitions per i-tile (one per wave)
#define JPART  (N_ROWS / JSPLIT)      // 1024 columns per partition

// ---------------------------------------------------------------------------
// Kernel 1: Wh = h @ W  (16x64 tile per wave, WMMA f16->f32),
//           fused Wh1 = Wh@a1, Wh2 = Wh@a2, and WhT (f16, [64][8192]) store.
// ---------------------------------------------------------------------------
__global__ __launch_bounds__(256) void gat_wh_kernel(
    const float* __restrict__ h, const float* __restrict__ W,
    const float* __restrict__ a, _Float16* __restrict__ WhT,
    float* __restrict__ Wh1, float* __restrict__ Wh2)
{
    __shared__ _Float16 WT[OUT_DIM * IN_DIM];  // [n][k] f16, 64KB

    const int tid = threadIdx.x;
    // Cooperative transpose of W [512,64] f32 -> WT [64][512] f16 in LDS
    #pragma unroll
    for (int i = 0; i < (OUT_DIM * IN_DIM) / 256; ++i) {
        int idx = tid + i * 256;
        int n = idx >> 9;        // / 512
        int k = idx & 511;       // % 512
        WT[n * IN_DIM + k] = (_Float16)W[k * OUT_DIM + n];
    }
    __syncthreads();

    const int wave = tid >> 5;
    const int lane = tid & 31;
    const int lr = lane & 15;    // 0..15
    const int lh = lane >> 4;    // 0..1
    const int r0 = (blockIdx.x * 8 + wave) * 16;

    v8f acc[4] = {};             // 16x64 f32 accumulator (4 N-tiles of 16)

    const float* hrow = h + (size_t)(r0 + lr) * IN_DIM;

    for (int ks = 0; ks < IN_DIM / 32; ++ks) {
        const int k0 = ks * 32 + lh * 16;
        // A fragment: row lr, K = k0..k0+15 (f32 -> f16)
        v16h afrag;
        const v4f* hp = (const v4f*)(hrow + k0);
        #pragma unroll
        for (int q = 0; q < 4; ++q) {
            v4f h4 = hp[q];
            #pragma unroll
            for (int c = 0; c < 4; ++c) afrag[q * 4 + c] = (_Float16)h4[c];
        }
        #pragma unroll
        for (int t = 0; t < 4; ++t) {
            // B fragment: col (16t+lr), K = k0..k0+15, contiguous in LDS
            v16h bfrag = *(const v16h*)(&WT[(t * 16 + lr) * IN_DIM + k0]);
            acc[t] = __builtin_amdgcn_wmma_f32_16x16x32_f16(
                false, afrag, false, bfrag, (short)0, acc[t], false, false);
        }
    }

    // Epilogue: store WhT (f16, column-major over rows) and reduce Wh@a1/a2
    float p1[8] = {}, p2[8] = {};
    #pragma unroll
    for (int t = 0; t < 4; ++t) {
        const int col = t * 16 + lr;
        const float a1 = a[col];
        const float a2 = a[OUT_DIM + col];
        #pragma unroll
        for (int v = 0; v < 8; ++v) {
            float val = acc[t][v];           // row r0 + v + 8*lh, col
            p1[v] += val * a1;
            p2[v] += val * a2;
            WhT[(size_t)col * N_ROWS + (r0 + v + lh * 8)] = (_Float16)val;
        }
    }
    #pragma unroll
    for (int v = 0; v < 8; ++v) {
        float s1 = p1[v], s2 = p2[v];
        #pragma unroll
        for (int m = 8; m >= 1; m >>= 1) {   // butterfly over lr (16 lanes)
            s1 += __shfl_xor(s1, m, 32);
            s2 += __shfl_xor(s2, m, 32);
        }
        if (lr == 0) {
            Wh1[r0 + v + lh * 8] = s1;
            Wh2[r0 + v + lh * 8] = s2;
        }
    }
}

// ---------------------------------------------------------------------------
// Kernel 2: fused masked-softmax attention + att@Wh + ELU.
// One block per 16-row i-tile; 8 waves each own a 1024-column j-partition
// (flash-attention split-K) -> 4096 waves for latency hiding on the 256 MB
// adj stream (read once, non-temporal so it doesn't evict the reused WhT/Wh2
// from L2). Partials merged in LDS (max + exp-rescale + ds_add_f32).
// ---------------------------------------------------------------------------
__global__ __launch_bounds__(256) void gat_attn_kernel(
    const int* __restrict__ adj, const _Float16* __restrict__ WhT,
    const float* __restrict__ Wh1, const float* __restrict__ Wh2,
    float* __restrict__ out)
{
    __shared__ float red[16 * OUT_DIM];   // merged numerator tile, 4KB
    __shared__ float ssum[16];            // merged row sums
    __shared__ float pmax[JSPLIT * 16];   // per-partition row maxima

    const int tid = threadIdx.x;
    const int wave = tid >> 5;            // j-partition index
    const int lane = tid & 31;
    const int lr = lane & 15;
    const int lh = lane >> 4;
    const int r0 = blockIdx.x * 16;
    const int row = r0 + lr;              // A-role row this lane owns

    const float wh1 = Wh1[row];
    const int* adjrow = adj + (size_t)row * N_ROWS;

    float m_run = NEGBIG, s_run = 0.0f;
    v8f acc[4] = {};

    const int jstart = wave * JPART;
    const int jend   = jstart + JPART;

    for (int j0 = jstart; j0 < jend; j0 += 32) {
        const int jb = j0 + lh * 16;

        if (j0 + 32 < jend)
            __builtin_prefetch(adjrow + jb + 32, 0, 0);

        // e_ij = leakyrelu(Wh1[i] + Wh2[j]); mask with adj
        int   av[16];
        float ev[16];
        float bmax = NEGBIG;
        const v4i* ap = (const v4i*)(adjrow + jb);
        const v4f* wp = (const v4f*)(Wh2 + jb);
        #pragma unroll
        for (int q = 0; q < 4; ++q) {
            v4i a4 = __builtin_nontemporal_load(ap + q);  // single-use stream
            v4f w4 = wp[q];
            #pragma unroll
            for (int c = 0; c < 4; ++c) {
                float e0 = wh1 + w4[c];
                e0 = e0 > 0.0f ? e0 : ALPHA * e0;
                const int jj = q * 4 + c;
                av[jj] = a4[c];
                ev[jj] = e0;
                if (a4[c] > 0) bmax = fmaxf(bmax, e0);
            }
        }
        bmax = fmaxf(bmax, __shfl_xor(bmax, 16, 32));   // pair halves of row
        const float m_new = fmaxf(m_run, bmax);
        const bool  any   = m_new > -1.0e29f;
        const float scale = any ? __expf(m_run - m_new) : 1.0f;

        // p fragment (A-matrix 16x32 f16 layout) + partial row sums
        float psum = 0.0f;
        v16h pfrag;
        #pragma unroll
        for (int jj = 0; jj < 16; ++jj) {
            float p = (any && av[jj] > 0) ? __expf(ev[jj] - m_new) : 0.0f;
            psum += p;
            pfrag[jj] = (_Float16)p;
        }
        psum += __shfl_xor(psum, 16, 32);
        s_run = s_run * scale + psum;
        m_run = m_new;

        // Rescale accumulator rows (C row v+8*lh lives in lane v+8*lh)
        #pragma unroll
        for (int v = 0; v < 8; ++v) {
            const float sc = __shfl(scale, v + 8 * lh, 32);
            #pragma unroll
            for (int t = 0; t < 4; ++t) acc[t][v] *= sc;
        }

        // acc += P(16x32) x Wh(32x64) via 4 WMMAs
        #pragma unroll
        for (int t = 0; t < 4; ++t) {
            const v16h bfrag =
                *(const v16h*)(WhT + (size_t)(t * 16 + lr) * N_ROWS + jb);
            acc[t] = __builtin_amdgcn_wmma_f32_16x16x32_f16(
                false, pfrag, false, bfrag, (short)0, acc[t], false, false);
        }
    }

    // ---- cross-partition merge in LDS ----
    // Publish per-partition row maxima; zero the merge buffers.
    if (lane < 16) pmax[wave * 16 + lane] = m_run;
    #pragma unroll
    for (int i = 0; i < 4; ++i) red[tid + i * 256] = 0.0f;
    if (tid < 16) ssum[tid] = 0.0f;
    __syncthreads();

    // Global row max over the 8 partitions (each lane: its row lr)
    float M = pmax[lr];
    #pragma unroll
    for (int w = 1; w < JSPLIT; ++w) M = fmaxf(M, pmax[w * 16 + lr]);

    // This partition's rescale factor for row lr (0 if row fully masked)
    const float factor = (M > -1.0e29f) ? __expf(m_run - M) : 0.0f;

    if (lh == 0) atomicAdd(&ssum[lr], factor * s_run);

    #pragma unroll
    for (int v = 0; v < 8; ++v) {
        const float f = __shfl(factor, v + 8 * lh, 32);   // factor for C row
        const int r = v + 8 * lh;
        #pragma unroll
        for (int t = 0; t < 4; ++t)
            atomicAdd(&red[r * OUT_DIM + t * 16 + lr], f * acc[t][v]);
    }
    __syncthreads();

    // Cooperative epilogue: divide by row sum, ELU, coalesced NT fp32 store
    #pragma unroll
    for (int i = 0; i < 4; ++i) {
        const int idx = tid + i * 256;          // 0..1023 over 16x64 tile
        const int r = idx >> 6;                 // row in tile
        const int c = idx & 63;                 // col
        const float s = ssum[r];
        const float inv = s > 0.0f ? 1.0f / s : 0.0f;
        float val = red[idx] * inv;
        val = val > 0.0f ? val : (__expf(val) - 1.0f);    // ELU (alpha=1)
        __builtin_nontemporal_store(val, &out[(size_t)(r0 + r) * OUT_DIM + c]);
    }
}

// ---------------------------------------------------------------------------
extern "C" void kernel_launch(void* const* d_in, const int* in_sizes, int n_in,
                              void* d_out, int out_size, void* d_ws, size_t ws_size,
                              hipStream_t stream) {
    const float* h   = (const float*)d_in[0];   // [8192, 512] f32
    const int*   adj = (const int*)  d_in[1];   // [8192, 8192] i32
    const float* W   = (const float*)d_in[2];   // [512, 64] f32
    const float* a   = (const float*)d_in[3];   // [128, 1] f32
    float* out = (float*)d_out;                 // [8192, 64] f32

    // Workspace layout: WhT f16 [64][8192] (1 MB), Wh1 f32 [8192], Wh2 f32 [8192]
    _Float16* WhT = (_Float16*)d_ws;
    float* Wh1 = (float*)((char*)d_ws + (size_t)OUT_DIM * N_ROWS * sizeof(_Float16));
    float* Wh2 = Wh1 + N_ROWS;

    // Kernel 1: 8192 rows / (16 rows per wave * 8 waves per block) = 64 blocks
    gat_wh_kernel<<<64, 256, 0, stream>>>(h, W, a, WhT, Wh1, Wh2);
    // Kernel 2: one block per 16-row i-tile, 8 j-partitions each
    gat_attn_kernel<<<512, 256, 0, stream>>>(adj, WhT, Wh1, Wh2, out);
}
